// StructuredTransformerDecoderBlock_13039520711126
// MI455X (gfx1250) — matrix operand order repaired and checked by reading, compiled, not run on hardware
//
#include <hip/hip_runtime.h>
#include <hip/hip_bf16.h>

// ---------------- problem constants ----------------
#define NN    16384
#define SZ    128
#define KNBR  32
#define EDGE  96
#define INS   224   // SZ + EDGE
#define HIDD  256
#define HEADS 8
#define AH    32

// ---------------- workspace offsets (in _Float16 units) ----------------
// f16 weight blocks (B-fragment layout), then localh, U, m buffers.
#define OFF_MW0   ((size_t)0)         // 128x256 -> 32768
#define OFF_MW1   ((size_t)32768)     // 256x256 -> 65536
#define OFF_MW2   ((size_t)98304)     // 256x128 -> 32768
#define OFF_WQ    ((size_t)131072)    // 128x256 -> 32768
#define OFF_WKT   ((size_t)163840)    // 8 heads x 14 nb x 512 -> 57344
#define OFF_WV    ((size_t)221184)    // 224x256 -> 57344
#define OFF_WO    ((size_t)278528)    // 256x128 -> 32768
#define OFF_LOCAL ((size_t)311296)    // NN*128 f16
#define OFF_U     ((size_t)2408448)   // NN*8*224 f16
#define OFF_M     ((size_t)31768576)  // NN*8*224 f16
// total 61128704 halfs = ~116.6 MB of d_ws

typedef __attribute__((ext_vector_type(16))) _Float16 v16h;
typedef __attribute__((ext_vector_type(8)))  _Float16 v8h;
typedef __attribute__((ext_vector_type(4)))  _Float16 v4h;
typedef __attribute__((ext_vector_type(8)))  float    v8f;

// ---------------- helpers ----------------
__device__ __forceinline__ v16h frag_cat(v8h lo, v8h hi) {
  return __builtin_shufflevector(lo, hi, 0,1,2,3,4,5,6,7,8,9,10,11,12,13,14,15);
}
__device__ __forceinline__ v8f wmma_f16(v16h a, v16h b, v8f c) {
  return __builtin_amdgcn_wmma_f32_16x16x32_f16(false, a, false, b, (short)0, c, false, false);
}
__device__ __forceinline__ v8f v8f_zero() {
  v8f z;
  #pragma unroll
  for (int i = 0; i < 8; ++i) z[i] = 0.0f;
  return z;
}
__device__ __forceinline__ v8h v8h_zero() {
  v8h z;
  #pragma unroll
  for (int i = 0; i < 8; ++i) z[i] = (_Float16)0;
  return z;
}
__device__ __forceinline__ float wsum32(float v) {
  #pragma unroll
  for (int m = 16; m >= 1; m >>= 1) v += __shfl_xor(v, m, 32);
  return v;
}
__device__ __forceinline__ float wmax32(float v) {
  #pragma unroll
  for (int m = 16; m >= 1; m >>= 1) v = fmaxf(v, __shfl_xor(v, m, 32));
  return v;
}

// A fragment (16x32 f16) from row-major f16 buffer (ld in halfs).
// layout: lanes 0-15 M=0..15 K=0..15; lanes 16-31 K=16..31
__device__ __forceinline__ v16h load_a_rm(const _Float16* base, int ld, int m0, int k0) {
  int lane = threadIdx.x & 31;
  int ko = lane >> 4;
  int m = m0 + (lane & 15);
  const _Float16* p = base + (size_t)m * ld + k0 + ko * 16;
  v8h lo = *(const v8h*)p;          // K = 16*ko + 0..7
  v8h hi = *(const v8h*)(p + 8);    // K = 16*ko + 8..15
  return frag_cat(lo, hi);
}

// B fragment (32x16 f16) from pre-swizzled block layout: block = 512 halfs,
// lane-major: blocks[lane*16 + i] holds W[kb*32 + 16*(lane>>4) + i, nb*16 + (lane&15)]
__device__ __forceinline__ v16h load_b_blk(const _Float16* blocks, int kbc, int nb, int kb) {
  int lane = threadIdx.x & 31;
  const _Float16* p = blocks + ((size_t)(nb * kbc + kb)) * 512 + lane * 16;
  v8h lo = *(const v8h*)p;
  v8h hi = *(const v8h*)(p + 8);
  return frag_cat(lo, hi);
}

// B fragment from a K-major-transposed buffer: buf[n][k] row-major (ld halfs).
// element i <-> k_local = 16*(lane>>4) + i, n = n0 + (lane&15)
__device__ __forceinline__ v16h load_b_tr(const _Float16* buf, int ld, int n0, int k0) {
  int lane = threadIdx.x & 31;
  int ko = lane >> 4;
  int n = n0 + (lane & 15);
  const _Float16* p = buf + (size_t)n * ld + k0 + ko * 16;
  v8h lo = *(const v8h*)p;
  v8h hi = *(const v8h*)(p + 8);
  return frag_cat(lo, hi);
}

// B fragment from row-major [K][N] buffer (strided scalar loads).
__device__ __forceinline__ v16h load_b_rm(const _Float16* buf, int ld, int n0, int k0) {
  int lane = threadIdx.x & 31;
  int ko = lane >> 4;
  int n = n0 + (lane & 15);
  const _Float16* p = buf + (size_t)(k0 + ko * 16) * ld + n;
  v16h b;
  #pragma unroll
  for (int i = 0; i < 16; ++i) b[i] = p[(size_t)i * ld];
  return b;
}

// GEMM: A (f16, row-major in LDS) x B (block layout in global) + bias -> f16 out
__device__ __forceinline__ void gemm_f16out(const _Float16* A, int lda,
                                            const _Float16* B, int kbc,
                                            const float* bias, bool relu,
                                            _Float16* out, int ldo,
                                            int mtile, int nt0, int nt1) {
  int lane = threadIdx.x & 31;
  int ko = lane >> 4, nin = lane & 15;
  for (int nt = nt0; nt < nt1; nt += 2) {
    v8f a0 = v8f_zero(), a1 = v8f_zero();
    for (int kb = 0; kb < kbc; ++kb) {
      __builtin_prefetch(B + ((size_t)(nt * kbc + kb + 1)) * 512, 0, 3);
      v16h a = load_a_rm(A, lda, mtile * 16, kb * 32);
      a0 = wmma_f16(a, load_b_blk(B, kbc, nt, kb), a0);
      a1 = wmma_f16(a, load_b_blk(B, kbc, nt + 1, kb), a1);
    }
    float b0 = bias[nt * 16 + nin], b1 = bias[(nt + 1) * 16 + nin];
    #pragma unroll
    for (int r = 0; r < 8; ++r) {
      int m = mtile * 16 + r + 8 * ko;
      float v0 = a0[r] + b0, v1 = a1[r] + b1;
      if (relu) { v0 = fmaxf(v0, 0.f); v1 = fmaxf(v1, 0.f); }
      out[(size_t)m * ldo + nt * 16 + nin] = (_Float16)v0;
      out[(size_t)m * ldo + (nt + 1) * 16 + nin] = (_Float16)v1;
    }
  }
}

// Same, but f32 output (for pre-LayerNorm buffer)
__device__ __forceinline__ void gemm_f32out(const _Float16* A, int lda,
                                            const _Float16* B, int kbc,
                                            const float* bias,
                                            float* out, int ldo,
                                            int mtile, int nt0, int nt1) {
  int lane = threadIdx.x & 31;
  int ko = lane >> 4, nin = lane & 15;
  for (int nt = nt0; nt < nt1; nt += 2) {
    v8f a0 = v8f_zero(), a1 = v8f_zero();
    for (int kb = 0; kb < kbc; ++kb) {
      v16h a = load_a_rm(A, lda, mtile * 16, kb * 32);
      a0 = wmma_f16(a, load_b_blk(B, kbc, nt, kb), a0);
      a1 = wmma_f16(a, load_b_blk(B, kbc, nt + 1, kb), a1);
    }
    float b0 = bias[nt * 16 + nin], b1 = bias[(nt + 1) * 16 + nin];
    #pragma unroll
    for (int r = 0; r < 8; ++r) {
      int m = mtile * 16 + r + 8 * ko;
      out[(size_t)m * ldo + nt * 16 + nin] = a0[r] + b0;
      out[(size_t)m * ldo + (nt + 1) * 16 + nin] = a1[r] + b1;
    }
  }
}

// ---------------- kernel 0: weight prep (f32 -> f16 B-fragment blocks) ----------------
__device__ __forceinline__ void write_blocks(const float* W, int Kd, int Nd,
                                             _Float16* dst, int t) {
  int block = t >> 9, within = t & 511;
  int lane = within >> 4, i = within & 15;
  int kbc = Kd >> 5;
  int nb = block / kbc, kb = block - nb * kbc;
  int k = kb * 32 + ((lane >> 4) << 4) + i;
  int nncol = nb * 16 + (lane & 15);
  dst[t] = (_Float16)W[(size_t)k * Nd + nncol];
}

__global__ __launch_bounds__(256) void prep_weights(
    const float* mW0, const float* mW1, const float* mW2,
    const float* Wq, const float* Wk, const float* Wv, const float* Wo,
    _Float16* ws) {
  int tid = blockIdx.x * 256 + threadIdx.x;
  if (tid < 32768)        write_blocks(mW0, 128, 256, ws + OFF_MW0, tid);
  else if (tid < 98304)   write_blocks(mW1, 256, 256, ws + OFF_MW1, tid - 32768);
  else if (tid < 131072)  write_blocks(mW2, 256, 128, ws + OFF_MW2, tid - 98304);
  else if (tid < 163840)  write_blocks(Wq, 128, 256, ws + OFF_WQ, tid - 131072);
  else if (tid < 221184) {
    // WkT blocks: B[a, c] = Wk[c, h*32 + a], per head: 14 blocks, K-dim = 32
    int t = tid - 163840;
    int block = t >> 9, within = t & 511;
    int lane = within >> 4, i = within & 15;
    int h = block / 14, nb = block - h * 14;
    int a = ((lane >> 4) << 4) + i;         // k_local 0..31
    int c = nb * 16 + (lane & 15);          // 0..223
    ws[OFF_WKT + t] = (_Float16)Wk[(size_t)c * 256 + h * 32 + a];
  }
  else if (tid < 278528)  write_blocks(Wv, 224, 256, ws + OFF_WV, tid - 221184);
  else if (tid < 311296)  write_blocks(Wo, 256, 128, ws + OFF_WO, tid - 278528);
}

// ---------------- kernel 1: LN + MLP + LN + q + U (64 rows/block, 8 waves) ----------------
__global__ __launch_bounds__(256) void node_kernel(
    const float* features, const float* g_bn, const float* b_bn,
    const float* g_ln, const float* b_ln,
    const float* mb0, const float* mb1, const float* mb2, const float* bq,
    _Float16* ws) {
  __shared__ __align__(16) unsigned char smem[65536];
  _Float16* xbuf = (_Float16*)(smem);             // [64][128] 16KB
  _Float16* h1   = (_Float16*)(smem + 32768);     // [64][256] 32KB
  _Float16* h2   = (_Float16*)(smem);             // [64][256] 32KB (over xbuf)
  float*    obuf = (float*)(smem + 32768);        // [64][128] f32 32KB (over h1)
  _Float16* lbuf = (_Float16*)(smem);             // [64][128] 16KB (over h2)
  _Float16* qbuf = (_Float16*)(smem + 32768);     // [64][256] 32KB (over obuf)

  const _Float16* bMW0 = ws + OFF_MW0;
  const _Float16* bMW1 = ws + OFF_MW1;
  const _Float16* bMW2 = ws + OFF_MW2;
  const _Float16* bWQ  = ws + OFF_WQ;
  const _Float16* bWKT = ws + OFF_WKT;
  _Float16* localh = ws + OFF_LOCAL;
  _Float16* U      = ws + OFF_U;

  int tid = threadIdx.x, w = tid >> 5, lane = tid & 31;
  int rbase = blockIdx.x * 64;

  // --- LayerNorm(features) + ReLU -> xbuf (f16) ---
  for (int it = 0; it < 8; ++it) {
    int rl = w * 8 + it;
    int row = rbase + rl;
    float4 xv = ((const float4*)(features + (size_t)row * SZ))[lane];
    float mu = wsum32(xv.x + xv.y + xv.z + xv.w) * (1.0f / SZ);
    float dx = xv.x - mu, dy = xv.y - mu, dz = xv.z - mu, dw = xv.w - mu;
    float var = wsum32(dx*dx + dy*dy + dz*dz + dw*dw) * (1.0f / SZ);
    float rs = rsqrtf(var + 1e-5f);
    float4 g4 = ((const float4*)g_bn)[lane];
    float4 b4 = ((const float4*)b_bn)[lane];
    v4h hv;
    hv[0] = (_Float16)fmaxf(dx * rs * g4.x + b4.x, 0.f);
    hv[1] = (_Float16)fmaxf(dy * rs * g4.y + b4.y, 0.f);
    hv[2] = (_Float16)fmaxf(dz * rs * g4.z + b4.z, 0.f);
    hv[3] = (_Float16)fmaxf(dw * rs * g4.w + b4.w, 0.f);
    *(v4h*)&xbuf[rl * SZ + lane * 4] = hv;
  }
  __syncthreads();

  int mtile = w & 3;
  int half8 = (w >> 2);

  // --- h1 = relu(x @ mW0 + mb0) : [64,128]x[128,256] ---
  gemm_f16out(xbuf, 128, bMW0, 4, mb0, true, h1, 256, mtile, half8 * 8, half8 * 8 + 8);
  __syncthreads();
  // --- h2 = relu(h1 @ mW1 + mb1) : [64,256]x[256,256] ---
  gemm_f16out(h1, 256, bMW1, 8, mb1, true, h2, 256, mtile, half8 * 8, half8 * 8 + 8);
  __syncthreads();
  // --- obuf = h2 @ mW2 + mb2 : [64,256]x[256,128] (f32) ---
  gemm_f32out(h2, 256, bMW2, 8, mb2, obuf, 128, mtile, half8 * 4, half8 * 4 + 4);
  __syncthreads();

  // --- local = relu(LayerNorm(obuf)) -> lbuf (f16) + localh (global f16) ---
  for (int it = 0; it < 8; ++it) {
    int rl = w * 8 + it;
    int row = rbase + rl;
    float4 xv = *(float4*)&obuf[rl * SZ + lane * 4];
    float mu = wsum32(xv.x + xv.y + xv.z + xv.w) * (1.0f / SZ);
    float dx = xv.x - mu, dy = xv.y - mu, dz = xv.z - mu, dw = xv.w - mu;
    float var = wsum32(dx*dx + dy*dy + dz*dz + dw*dw) * (1.0f / SZ);
    float rs = rsqrtf(var + 1e-5f);
    float4 g4 = ((const float4*)g_ln)[lane];
    float4 b4 = ((const float4*)b_ln)[lane];
    v4h hv;
    hv[0] = (_Float16)fmaxf(dx * rs * g4.x + b4.x, 0.f);
    hv[1] = (_Float16)fmaxf(dy * rs * g4.y + b4.y, 0.f);
    hv[2] = (_Float16)fmaxf(dz * rs * g4.z + b4.z, 0.f);
    hv[3] = (_Float16)fmaxf(dw * rs * g4.w + b4.w, 0.f);
    *(v4h*)&lbuf[rl * SZ + lane * 4] = hv;
    *(v4h*)&localh[(size_t)row * SZ + lane * 4] = hv;
  }
  __syncthreads();

  // --- q = local @ Wq + bq : [64,128]x[128,256] -> qbuf f16 ---
  gemm_f16out(lbuf, 128, bWQ, 4, bq, false, qbuf, 256, mtile, half8 * 8, half8 * 8 + 8);
  __syncthreads();

  // --- U[n,h,:] = q[n,h,:] @ Wk_h^T  (per head, K=32) -> global f16 ---
  int ko = lane >> 4, nin = lane & 15;
  for (int hh = half8 * 4; hh < half8 * 4 + 4; ++hh) {
    v16h a = load_a_rm(qbuf, 256, mtile * 16, hh * 32);
    const _Float16* bh = bWKT + (size_t)hh * 14 * 512;
    for (int nb = 0; nb < 14; ++nb) {
      v8f acc = v8f_zero();
      acc = wmma_f16(a, load_b_blk(bh, 1, nb, 0), acc);
      #pragma unroll
      for (int r = 0; r < 8; ++r) {
        int row = rbase + mtile * 16 + r + 8 * ko;
        U[(size_t)row * (HEADS * INS) + hh * INS + nb * 16 + nin] = (_Float16)acc[r];
      }
    }
  }
}

// ---------------- kernel 2: per-node attention (2 nodes/block, wave per node) ----------------
__global__ __launch_bounds__(64) void attn_kernel(
    const int* neighbours, const float* edge_feats, _Float16* ws) {
  __shared__ __align__(16) _Float16 kvS[2][KNBR][INS];   // 28672 B
  __shared__ __align__(16) _Float16 uS[2][16][INS];      // 14336 B (rows 8..15 zero)
  __shared__ __align__(16) float    lgS[2][KNBR][16];    // 4096 B
  __shared__ __align__(16) _Float16 atS[2][16][KNBR];    // 2048 B

  const _Float16* localh = ws + OFF_LOCAL;
  const _Float16* U      = ws + OFF_U;
  _Float16* Mws          = ws + OFF_M;

  int lane = threadIdx.x & 31;
  int w = threadIdx.x >> 5;
  int n = blockIdx.x * 2 + w;
  int ko = lane >> 4, nin = lane & 15;

  // gather neighbour local features (f16, L2-resident)
  int idx = neighbours[(size_t)n * KNBR + lane];
  const _Float16* src = localh + (size_t)idx * SZ;
  #pragma unroll
  for (int i = 0; i < 16; ++i)
    *(v8h*)&kvS[w][lane][i * 8] = *(const v8h*)(src + i * 8);

  // edge features f32 -> f16
  const float* ef = edge_feats + ((size_t)n * KNBR + lane) * EDGE;
  #pragma unroll
  for (int i = 0; i < 12; ++i) {
    float4 e0 = ((const float4*)ef)[i * 2];
    float4 e1 = ((const float4*)ef)[i * 2 + 1];
    v8h hv;
    hv[0] = (_Float16)e0.x; hv[1] = (_Float16)e0.y;
    hv[2] = (_Float16)e0.z; hv[3] = (_Float16)e0.w;
    hv[4] = (_Float16)e1.x; hv[5] = (_Float16)e1.y;
    hv[6] = (_Float16)e1.z; hv[7] = (_Float16)e1.w;
    *(v8h*)&kvS[w][lane][SZ + i * 8] = hv;
  }

  // stage U[n] (8x224) + zero pad rows 8..15
  const _Float16* un = U + (size_t)n * (HEADS * INS);
  v8h z8 = v8h_zero();
  #pragma unroll
  for (int i = 0; i < 7; ++i) {
    int off = (i * 32 + lane) * 8;   // 0..1784 halfs
    *(v8h*)((_Float16*)&uS[w][0][0] + off) = *(const v8h*)(un + off);
    *(v8h*)((_Float16*)&uS[w][0][0] + HEADS * INS + off) = z8;
  }

  // logits[k,h] = sum_c kv[k,c] * U[h,c], scaled
  const float sc = 0.17677669529663687f;  // 1/sqrt(32)
  #pragma unroll
  for (int mt = 0; mt < 2; ++mt) {
    v8f acc = v8f_zero();
    for (int kc = 0; kc < 7; ++kc) {
      v16h a = load_a_rm(&kvS[w][0][0], INS, mt * 16, kc * 32);
      v16h b = load_b_tr(&uS[w][0][0], INS, 0, kc * 32);
      acc = wmma_f16(a, b, acc);
    }
    #pragma unroll
    for (int r = 0; r < 8; ++r)
      lgS[w][mt * 16 + r + 8 * ko][nin] = acc[r] * sc;
  }

  // softmax over K=32 per head (lane = neighbour index)
  #pragma unroll
  for (int h = 0; h < HEADS; ++h) {
    float s = lgS[w][lane][h];
    float mx = wmax32(s);
    float e = __expf(s - mx);
    float sm = wsum32(e);
    atS[w][h][lane] = (_Float16)(e / sm);
  }
  #pragma unroll
  for (int h = HEADS; h < 16; ++h) atS[w][h][lane] = (_Float16)0;

  // m[h,c] = sum_k attn[h,k] * kv[k,c]  (A = attn 16x32, B = kv 32x16 chunks)
  v16h aA = load_a_rm(&atS[w][0][0], KNBR, 0, 0);
  for (int cb = 0; cb < 14; ++cb) {
    v16h b = load_b_rm(&kvS[w][0][0], INS, cb * 16, 0);
    v8f acc = v8f_zero();
    acc = wmma_f16(aA, b, acc);
    if (ko == 0) {  // rows h = r (0..7) valid
      #pragma unroll
      for (int r = 0; r < 8; ++r)
        Mws[(size_t)n * (HEADS * INS) + r * INS + cb * 16 + nin] = (_Float16)acc[r];
    }
  }
}

// ---------------- kernel 3: ctx = m@Wv + bv ; out = ctx@Wo + bo + features ----------------
__global__ __launch_bounds__(256) void out_kernel(
    const float* features, const float* bv, const float* bo,
    _Float16* ws, float* out) {
  __shared__ __align__(16) _Float16 ctx[128 * 256];  // 64KB

  const _Float16* Mws = ws + OFF_M;
  const _Float16* bWV = ws + OFF_WV;
  const _Float16* bWO = ws + OFF_WO;

  int w = threadIdx.x >> 5, lane = threadIdx.x & 31;
  int ko = lane >> 4, nin = lane & 15;
  int rbase = blockIdx.x * 128;

  // ctx[:, h*32+a] = M_h @ Wv[:, h*32+a] + bv
  for (int nt = 0; nt < 16; ++nt) {
    int h = nt >> 1;
    v8f acc = v8f_zero();
    const _Float16* Abase = Mws + (size_t)rbase * (HEADS * INS) + h * INS;
    for (int kb = 0; kb < 7; ++kb) {
      v16h a = load_a_rm(Abase, HEADS * INS, w * 16, kb * 32);
      v16h b = load_b_blk(bWV, 7, nt, kb);
      acc = wmma_f16(a, b, acc);
    }
    float bb = bv[nt * 16 + nin];
    #pragma unroll
    for (int r = 0; r < 8; ++r)
      ctx[(size_t)(w * 16 + r + 8 * ko) * 256 + nt * 16 + nin] = (_Float16)(acc[r] + bb);
  }
  __syncthreads();

  // out = ctx @ Wo + bo + features
  for (int nt = 0; nt < 8; ++nt) {
    v8f acc = v8f_zero();
    for (int kb = 0; kb < 8; ++kb) {
      v16h a = load_a_rm(ctx, 256, w * 16, kb * 32);
      v16h b = load_b_blk(bWO, 8, nt, kb);
      acc = wmma_f16(a, b, acc);
    }
    int col = nt * 16 + nin;
    float bb = bo[col];
    #pragma unroll
    for (int r = 0; r < 8; ++r) {
      int row = rbase + w * 16 + r + 8 * ko;
      out[(size_t)row * SZ + col] = acc[r] + bb + features[(size_t)row * SZ + col];
    }
  }
}

// ---------------- launch ----------------
extern "C" void kernel_launch(void* const* d_in, const int* in_sizes, int n_in,
                              void* d_out, int out_size, void* d_ws, size_t ws_size,
                              hipStream_t stream) {
  const float* features = (const float*)d_in[0];
  const int*   neigh    = (const int*)d_in[1];
  const float* edge     = (const float*)d_in[2];
  const float* g_bn = (const float*)d_in[3];
  const float* b_bn = (const float*)d_in[4];
  const float* g_ln = (const float*)d_in[5];
  const float* b_ln = (const float*)d_in[6];
  const float* mW0 = (const float*)d_in[7];
  const float* mb0 = (const float*)d_in[8];
  const float* mW1 = (const float*)d_in[9];
  const float* mb1 = (const float*)d_in[10];
  const float* mW2 = (const float*)d_in[11];
  const float* mb2 = (const float*)d_in[12];
  const float* Wq = (const float*)d_in[13];
  const float* bq = (const float*)d_in[14];
  const float* Wk = (const float*)d_in[15];
  // d_in[16] = bk : mathematically cancels in softmax, unused
  const float* Wv = (const float*)d_in[17];
  const float* bv = (const float*)d_in[18];
  const float* Wo = (const float*)d_in[19];
  const float* bo = (const float*)d_in[20];

  _Float16* ws = (_Float16*)d_ws;
  float* outp = (float*)d_out;

  prep_weights<<<1216, 256, 0, stream>>>(mW0, mW1, mW2, Wq, Wk, Wv, Wo, ws);
  node_kernel<<<NN / 64, 256, 0, stream>>>(features, g_bn, b_bn, g_ln, b_ln,
                                           mb0, mb1, mb2, bq, ws);
  attn_kernel<<<NN / 2, 64, 0, stream>>>(neigh, edge, ws);
  out_kernel<<<NN / 128, 256, 0, stream>>>(features, bv, bo, ws, outp);
}
